// EqMotion_74869869904618
// MI455X (gfx1250) — matrix operand order repaired and checked by reading, not compiled, not run on hardware
//
#include <hip/hip_runtime.h>
#include <math.h>

#define BB 4
#define NN 256
#define TIN 10
#define TOUT 20
#define PI_F 3.14159265358979323846f

typedef __attribute__((ext_vector_type(16))) _Float16 v16h;
typedef __attribute__((ext_vector_type(8)))  float    v8f;

// ---------------- workspace layout (float region, then half region) ----------------
#define F_XC    0                         // 16 floats (B*4): x_center sums
#define F_XM    16                        // 16: x_mean sums (post-DCT)
#define F_XM2   32                        // 16: xh mean sums
#define F_XDCT  64                        // B*N*10*3 = 30720
#define F_VDCT  (F_XDCT + 30720)          // 30720
#define F_COORD (F_VDCT + 30720)          // B*N*128*3 = 393216
#define F_HH    (F_COORD + 393216)        // B*N*64 = 65536
#define F_AGG   (F_HH + 65536)            // 65536
#define F_END   (F_AGG + 65536)

#define H_HH    0                          // B*N*64 halves
#define H_NODE  65536                      // B*N*64
#define H_WC1   131072                     // 128x64  -> 8192
#define H_WC2   (H_WC1 + 8192)             // 64x128  -> 8192
#define H_WE1   (H_WC2 + 8192)             // 256x64  -> 16384
#define H_WE2   (H_WE1 + 16384)            // 64x64   -> 4096
#define H_WK1   (H_WE2 + 4096)             // 256x64  -> 16384
#define H_WK2   (H_WK1 + 16384)            // 64x16(pad) -> 1024

// ---------------- helpers ----------------
__device__ __forceinline__ void lds_fence() {
    asm volatile("s_wait_dscnt 0" ::: "memory");
}

// fast SiLU: v_exp_f32 + v_rcp_f32 + v_mul (avoid IEEE div_scale/div_fixup chain)
__device__ __forceinline__ float silu_f(float x) {
    return x * __builtin_amdgcn_rcpf(1.f + __expf(-x));
}

__device__ __forceinline__ v8f wmma32(v16h a, v16h b, v8f c) {
    return __builtin_amdgcn_wmma_f32_16x16x32_f16(false, a, false, b, (short)0, c, false, false);
}

// A-fragment (16x32 f16) from LDS row-major tile [16][ld]
__device__ __forceinline__ v16h afrag(const _Float16* buf, int ld, int k0, int lane) {
    int row = lane & 15;
    int b1  = (lane < 16) ? 0 : 8;
    const _Float16* p = buf + row * ld + k0 + b1;
    v16h a;
#pragma unroll
    for (int h = 0; h < 8; ++h) { a[h] = p[h]; a[8 + h] = p[16 + h]; }
    return a;
}

// A-fragment where every row is the same vector (broadcast node-i features)
__device__ __forceinline__ v16h afrag_bcast(const _Float16* vec, int k0, int lane) {
    int b1 = (lane < 16) ? 0 : 8;
    const _Float16* p = vec + k0 + b1;
    v16h a;
#pragma unroll
    for (int h = 0; h < 8; ++h) { a[h] = p[h]; a[8 + h] = p[16 + h]; }
    return a;
}

// B-fragment from pre-swizzled weights: [frag][lane][16] halves, contiguous per lane
__device__ __forceinline__ v16h bfrag(const _Float16* frags, int frag, int lane) {
    return *(const v16h*)(frags + ((size_t)((frag << 5) + lane) << 4));
}

// write D tile (+bias, SiLU) as f16 into LDS row-major [16][ld]
__device__ __forceinline__ void store_act(_Float16* buf, int ld, int c0, int lane, v8f acc, float bias) {
    int base8 = (lane < 16) ? 0 : 8;
    int col   = c0 + (lane & 15);
#pragma unroll
    for (int v = 0; v < 8; ++v)
        buf[(base8 + v) * ld + col] = (_Float16)silu_f(acc[v] + bias);
}

__device__ __forceinline__ float dctc(int k, int i, int n) {
    float w = (k == 0) ? sqrtf(1.f / n) : sqrtf(2.f / n);
    return w * cosf(PI_F * (i + 0.5f) * k / n);
}

// dist tile (16 rows j0..j0+15 x 128 channels) -> coord_dist MLP -> f16 in sD
__device__ void cd_tile(int lane, int b, int i, int j0, const float* coord,
                        const float* sCi, _Float16* sD, _Float16* sA1,
                        const _Float16* Wc1s, const _Float16* Wc2s,
                        const float* bc1, const float* bc2) {
    for (int idx = lane; idx < 16 * 128; idx += 32) {
        int r = idx >> 7, c = idx & 127;
        const float* cj = coord + (size_t)((b * NN + j0 + r) * 128 + c) * 3;
        float dx = sCi[c * 3 + 0] - cj[0];
        float dy = sCi[c * 3 + 1] - cj[1];
        float dz = sCi[c * 3 + 2] - cj[2];
        sD[idx] = (_Float16)sqrtf(dx * dx + dy * dy + dz * dz);
    }
    lds_fence();
    // layer1: [16x128] x [128x64]
    for (int ct = 0; ct < 4; ++ct) {
        v8f acc = {};
#pragma unroll
        for (int kt = 0; kt < 4; ++kt)
            acc = wmma32(afrag(sD, 128, kt * 32, lane), bfrag(Wc1s, kt * 4 + ct, lane), acc);
        store_act(sA1, 64, ct * 16, lane, acc, bc1[ct * 16 + (lane & 15)]);
    }
    lds_fence();
    // layer2: [16x64] x [64x128] -> overwrite sD with coord_dist
    for (int ct = 0; ct < 8; ++ct) {
        v8f acc = {};
#pragma unroll
        for (int kt = 0; kt < 2; ++kt)
            acc = wmma32(afrag(sA1, 64, kt * 32, lane), bfrag(Wc2s, kt * 8 + ct, lane), acc);
        store_act(sD, 128, ct * 16, lane, acc, bc2[ct * 16 + (lane & 15)]);
    }
    lds_fence();
}

// layer with A = [hi(64) | hj(64) | cd(128)], W [256x64] -> SiLU -> sA1
__device__ void edge_l1(int lane, const _Float16* sHi, const _Float16* sHj, const _Float16* sD,
                        _Float16* sA1, const _Float16* W1s, const float* b1) {
    for (int ct = 0; ct < 4; ++ct) {
        v8f acc = {};
#pragma unroll
        for (int kt = 0; kt < 2; ++kt)
            acc = wmma32(afrag_bcast(sHi, kt * 32, lane), bfrag(W1s, kt * 4 + ct, lane), acc);
#pragma unroll
        for (int kt = 0; kt < 2; ++kt)
            acc = wmma32(afrag(sHj, 64, kt * 32, lane), bfrag(W1s, (2 + kt) * 4 + ct, lane), acc);
#pragma unroll
        for (int kt = 0; kt < 4; ++kt)
            acc = wmma32(afrag(sD, 128, kt * 32, lane), bfrag(W1s, (4 + kt) * 4 + ct, lane), acc);
        store_act(sA1, 64, ct * 16, lane, acc, b1[ct * 16 + (lane & 15)]);
    }
    lds_fence();
}

// ---------------- prep kernels ----------------
__global__ void k_zero(float* ws) {
    int t = blockIdx.x * blockDim.x + threadIdx.x;
    if (t < 48) ws[F_XC + t] = 0.f;
    if (t < 65536) ws[F_AGG + t] = 0.f;
}

__global__ void k_center(const float* x, float* ws) {
    int t = blockIdx.x * 256 + threadIdx.x;
    int b = t >> 8, n = t & 255;
    const float* xp = x + (size_t)(b * NN + n) * TIN * 3;
    float s0 = 0, s1 = 0, s2 = 0;
    for (int tt = 0; tt < TIN; ++tt) { s0 += xp[tt * 3]; s1 += xp[tt * 3 + 1]; s2 += xp[tt * 3 + 2]; }
    atomicAdd(&ws[F_XC + b * 4 + 0], s0);
    atomicAdd(&ws[F_XC + b * 4 + 1], s1);
    atomicAdd(&ws[F_XC + b * 4 + 2], s2);
}

__global__ void k_dct_emb(const float* x, const float* vel, const float* h,
                          const float* W_emb, const float* b_emb,
                          const float* W_emb2, const float* b_emb2,
                          float* ws, _Float16* hhh) {
    int t = blockIdx.x * 256 + threadIdx.x;
    int b = t >> 8, n = t & 255;
    size_t node = (size_t)(b * NN + n);
    const float* xp = x + node * TIN * 3;
    const float* vp = vel + node * TIN * 3;
    float xc[3];
    for (int d = 0; d < 3; ++d) xc[d] = ws[F_XC + b * 4 + d] * (1.f / 2560.f);
    // velocity angle (pre-DCT)
    float ang[TIN];
    for (int tt = 0; tt < TIN; ++tt) {
        int tp = tt ? tt - 1 : 0;
        float dot = 0, na = 0, nb = 0;
        for (int d = 0; d < 3; ++d) {
            float a = vp[tp * 3 + d], bq = vp[tt * 3 + d];
            dot += a * bq; na += a * a; nb += bq * bq;
        }
        float ca = dot / ((sqrtf(na) + 1e-6f) * (sqrtf(nb) + 1e-6f));
        ca = fminf(1.f, fmaxf(-1.f, ca));
        ang[tt] = acosf(ca);
    }
    // DCT of centered x and of vel
    float sx[3] = {0, 0, 0};
    for (int s = 0; s < TIN; ++s) {
        for (int d = 0; d < 3; ++d) {
            float a = 0, v = 0;
            for (int tt = 0; tt < TIN; ++tt) {
                float c = dctc(s, tt, TIN);
                a += c * (xp[tt * 3 + d] - xc[d]);
                v += c * vp[tt * 3 + d];
            }
            ws[F_XDCT + (node * TIN + s) * 3 + d] = a;
            ws[F_VDCT + (node * TIN + s) * 3 + d] = v;
            sx[d] += a;
        }
    }
    for (int d = 0; d < 3; ++d) atomicAdd(&ws[F_XM + b * 4 + d], sx[d]);
    // node embedding hh = [h@W_emb+b | ang@W_emb2+b]
    const float* hp = h + node * TIN;
    for (int k = 0; k < 32; ++k) {
        float e1 = b_emb[k], e2 = b_emb2[k];
        for (int tt = 0; tt < TIN; ++tt) {
            e1 += hp[tt] * W_emb[tt * 32 + k];
            e2 += ang[tt] * W_emb2[tt * 32 + k];
        }
        ws[F_HH + node * 64 + k] = e1;
        ws[F_HH + node * 64 + 32 + k] = e2;
        hhh[node * 64 + k] = (_Float16)e1;
        hhh[node * 64 + 32 + k] = (_Float16)e2;
    }
}

__global__ void k_coord(const float* W_coord, const float* W_vel, float* ws) {
    int t = blockIdx.x * 256 + threadIdx.x;   // 65536 = B*N*64
    int c = t & 63;
    int node = t >> 6;
    int b = node >> 8;
    float xm[3];
    for (int d = 0; d < 3; ++d) xm[d] = ws[F_XM + b * 4 + d] * (1.f / 2560.f);
    const float* xd = ws + F_XDCT + (size_t)node * TIN * 3;
    const float* vd = ws + F_VDCT + (size_t)node * TIN * 3;
    float xh[3] = {0, 0, 0}, vh[3] = {0, 0, 0};
    for (int tt = 0; tt < TIN; ++tt) {
        float wc = W_coord[tt * 64 + c], wv = W_vel[tt * 64 + c];
        for (int d = 0; d < 3; ++d) {
            xh[d] += wc * (xd[tt * 3 + d] - xm[d]);
            vh[d] += wv * vd[tt * 3 + d];
        }
    }
    for (int d = 0; d < 3; ++d) {
        float val = xh[d] + xm[d];
        ws[F_COORD + ((size_t)node * 128 + c) * 3 + d] = val;
        ws[F_COORD + ((size_t)node * 128 + 64 + c) * 3 + d] = vh[d];
        atomicAdd(&ws[F_XM2 + b * 4 + d], val);
    }
}

__global__ void k_xout(const float* W_pred, const float* ws, float* out) {
    int t = blockIdx.x * 256 + threadIdx.x;   // 1024 nodes
    int node = t, b = t >> 8;
    float xm2[3], xc[3];
    for (int d = 0; d < 3; ++d) {
        xm2[d] = ws[F_XM2 + b * 4 + d] * (1.f / 16384.f);
        xc[d]  = ws[F_XC + b * 4 + d] * (1.f / 2560.f);
    }
    float xp2[TOUT][3];
    for (int o = 0; o < TOUT; ++o)
        for (int d = 0; d < 3; ++d) {
            float s = xm2[d];
            for (int c = 0; c < 64; ++c)
                s += W_pred[c * TOUT + o] * (ws[F_COORD + ((size_t)node * 128 + c) * 3 + d] - xm2[d]);
            xp2[o][d] = s;
        }
    for (int s = 0; s < TOUT; ++s)
        for (int d = 0; d < 3; ++d) {
            float a = xc[d];
            for (int o = 0; o < TOUT; ++o) a += dctc(o, s, TOUT) * xp2[o][d];  // idct = dct20^T
            out[((size_t)node * TOUT + s) * 3 + d] = a;
        }
}

// swizzle fp32 weight [K][Csrc] into WMMA B-fragment order (pad cols to 16*nct)
__global__ void k_swz(const float* W, _Float16* dst, int nkt, int nct, int Csrc) {
    int t = blockIdx.x * blockDim.x + threadIdx.x;
    if (t >= nkt * nct * 32) return;
    int lane = t & 31, frag = t >> 5;
    int kt = frag / nct, ct = frag % nct;
    int kb = kt * 32 + ((lane < 16) ? 0 : 8);
    int c  = ct * 16 + (lane & 15);
#pragma unroll
    for (int h = 0; h < 16; ++h) {
        float v = (c < Csrc) ? W[(kb + h + ((h >= 8) ? 8 : 0)) * Csrc + c] : 0.f;
        dst[((size_t)frag * 32 + lane) * 16 + h] = (_Float16)v;
    }
}

__global__ void k_node(const float* Wn1, const float* bn1, const float* Wn2, const float* bn2,
                       const float* ws, _Float16* nnh) {
    int t = blockIdx.x * 256 + threadIdx.x;  // 1024 nodes
    size_t node = (size_t)t;
    float in[128];
    for (int k = 0; k < 64; ++k) in[k] = ws[F_HH + node * 64 + k];
    for (int k = 0; k < 64; ++k) in[64 + k] = ws[F_AGG + node * 64 + k];
    float t1[64];
    for (int k = 0; k < 64; ++k) {
        float s = bn1[k];
        for (int j = 0; j < 128; ++j) s += in[j] * Wn1[j * 64 + k];
        t1[k] = silu_f(s);
    }
    for (int k = 0; k < 64; ++k) {
        float s = bn2[k];
        for (int j = 0; j < 64; ++j) s += t1[j] * Wn2[j * 64 + k];
        nnh[node * 64 + k] = (_Float16)silu_f(s);
    }
}

// ---------------- pass 1: edge_feat + masked aggregation (WMMA) ----------------
__global__ __launch_bounds__(128) void k_edge1(const float* coord, const _Float16* hhh,
        const _Float16* Wc1s, const _Float16* Wc2s, const _Float16* We1s, const _Float16* We2s,
        const float* bc1, const float* bc2, const float* be1, const float* be2,
        float* agg) {
    __shared__ _Float16 sD[4][16 * 128];
    __shared__ _Float16 sA1[4][16 * 64];
    __shared__ _Float16 sHj[4][16 * 64];
    __shared__ _Float16 sHi[64];
    __shared__ float sCi[128 * 3];
    __shared__ float sAgg[64];
    int b = blockIdx.x >> 8, i = blockIdx.x & 255;
    int w = threadIdx.x >> 5, lane = threadIdx.x & 31;
    for (int t = threadIdx.x; t < 384; t += 128) sCi[t] = coord[(size_t)(b * NN + i) * 384 + t];
    for (int t = threadIdx.x; t < 64; t += 128) sHi[t] = hhh[(size_t)(b * NN + i) * 64 + t];
    if (threadIdx.x < 64) sAgg[threadIdx.x] = 0.f;
    __syncthreads();

    float accAgg[4] = {0.f, 0.f, 0.f, 0.f};
    int base8 = (lane < 16) ? 0 : 8;
    for (int q = 0; q < 4; ++q) {
        int j0 = (w + q * 4) * 16;
        for (int t = lane; t < 1024; t += 32)
            sHj[w][t] = hhh[(size_t)(b * NN + j0 + (t >> 6)) * 64 + (t & 63)];
        cd_tile(lane, b, i, j0, coord, sCi, sD[w], sA1[w], Wc1s, Wc2s, bc1, bc2);
        edge_l1(lane, sHi, sHj[w], sD[w], sA1[w], We1s, be1);
        // edge layer2 [16x64]x[64x64] + SiLU + masked row-sum into agg accumulator
        for (int ct = 0; ct < 4; ++ct) {
            v8f acc = {};
#pragma unroll
            for (int kt = 0; kt < 2; ++kt)
                acc = wmma32(afrag(sA1[w], 64, kt * 32, lane), bfrag(We2s, kt * 4 + ct, lane), acc);
            float bias = be2[ct * 16 + (lane & 15)];
            float s = 0.f;
#pragma unroll
            for (int v = 0; v < 8; ++v) {
                float val = silu_f(acc[v] + bias);
                if (j0 + base8 + v != i) s += val;   // mask = 1 - eye(N)
            }
            s += __shfl_xor(s, 16, 32);              // combine row halves
            if (lane < 16) accAgg[ct] += s;
        }
        lds_fence();
    }
    if (lane < 16)
        for (int ct = 0; ct < 4; ++ct) atomicAdd(&sAgg[ct * 16 + lane], accAgg[ct]);
    __syncthreads();
    if (threadIdx.x < 64) agg[(size_t)(b * NN + i) * 64 + threadIdx.x] = sAgg[threadIdx.x];
}

// ---------------- pass 2: category logits + softmax (WMMA, cd recomputed) ----------------
__global__ __launch_bounds__(128) void k_edge2(const float* coord, const _Float16* nnh,
        const _Float16* Wc1s, const _Float16* Wc2s, const _Float16* Wk1s, const _Float16* Wk2s,
        const float* bc1, const float* bc2, const float* bk1, const float* bk2,
        float* cat_out) {
    __shared__ _Float16 sD[4][16 * 128];
    __shared__ _Float16 sA1[4][16 * 64];
    __shared__ _Float16 sHj[4][16 * 64];
    __shared__ _Float16 sHi[64];
    __shared__ float sCi[128 * 3];
    int b = blockIdx.x >> 8, i = blockIdx.x & 255;
    int w = threadIdx.x >> 5, lane = threadIdx.x & 31;
    for (int t = threadIdx.x; t < 384; t += 128) sCi[t] = coord[(size_t)(b * NN + i) * 384 + t];
    for (int t = threadIdx.x; t < 64; t += 128) sHi[t] = nnh[(size_t)(b * NN + i) * 64 + t];
    __syncthreads();

    int base8 = (lane < 16) ? 0 : 8;
    for (int q = 0; q < 4; ++q) {
        int j0 = (w + q * 4) * 16;
        for (int t = lane; t < 1024; t += 32)
            sHj[w][t] = nnh[(size_t)(b * NN + j0 + (t >> 6)) * 64 + (t & 63)];
        cd_tile(lane, b, i, j0, coord, sCi, sD[w], sA1[w], Wc1s, Wc2s, bc1, bc2);
        edge_l1(lane, sHi, sHj[w], sD[w], sA1[w], Wk1s, bk1);
        // logits layer [16x64]x[64x16(pad)] — only cols 0..3 valid
        v8f acc = {};
#pragma unroll
        for (int kt = 0; kt < 2; ++kt)
            acc = wmma32(afrag(sA1[w], 64, kt * 32, lane), bfrag(Wk2s, kt, lane), acc);
        int cidx = lane & 15;
        float bias = (cidx < 4) ? bk2[cidx] : 0.f;
        float* sLog = (float*)sD[w];    // cd is dead; reuse as 16x16 f32 logits
#pragma unroll
        for (int v = 0; v < 8; ++v) sLog[(base8 + v) * 16 + cidx] = acc[v] + bias;
        lds_fence();
        if (lane < 16) {
            const float* L = sLog + lane * 16;
            float m = fmaxf(fmaxf(L[0], L[1]), fmaxf(L[2], L[3]));
            float e0 = __expf(L[0] - m), e1 = __expf(L[1] - m);
            float e2 = __expf(L[2] - m), e3 = __expf(L[3] - m);
            float inv = __builtin_amdgcn_rcpf(e0 + e1 + e2 + e3);
            float* o = cat_out + (size_t)(((b * NN + i) * NN) + j0 + lane) * 4;
            o[0] = e0 * inv; o[1] = e1 * inv; o[2] = e2 * inv; o[3] = e3 * inv;
        }
        lds_fence();
    }
}

// ---------------- launcher ----------------
extern "C" void kernel_launch(void* const* d_in, const int* in_sizes, int n_in,
                              void* d_out, int out_size, void* d_ws, size_t ws_size,
                              hipStream_t stream) {
    (void)in_sizes; (void)n_in; (void)out_size; (void)ws_size;
    const float* h      = (const float*)d_in[0];
    const float* x      = (const float*)d_in[1];
    const float* vel    = (const float*)d_in[2];
    const float* W_emb  = (const float*)d_in[3];
    const float* b_emb  = (const float*)d_in[4];
    const float* W_emb2 = (const float*)d_in[5];
    const float* b_emb2 = (const float*)d_in[6];
    const float* W_coord= (const float*)d_in[7];
    const float* W_vel  = (const float*)d_in[8];
    const float* W_pred = (const float*)d_in[9];
    const float* Wc1 = (const float*)d_in[10]; const float* bc1 = (const float*)d_in[11];
    const float* Wc2 = (const float*)d_in[12]; const float* bc2 = (const float*)d_in[13];
    const float* We1 = (const float*)d_in[14]; const float* be1 = (const float*)d_in[15];
    const float* We2 = (const float*)d_in[16]; const float* be2 = (const float*)d_in[17];
    const float* Wn1 = (const float*)d_in[18]; const float* bn1 = (const float*)d_in[19];
    const float* Wn2 = (const float*)d_in[20]; const float* bn2 = (const float*)d_in[21];
    const float* Wk1 = (const float*)d_in[22]; const float* bk1 = (const float*)d_in[23];
    const float* Wk2 = (const float*)d_in[24]; const float* bk2 = (const float*)d_in[25];

    float* WS = (float*)d_ws;
    _Float16* HS = (_Float16*)((char*)d_ws + (size_t)F_END * 4);
    float* out   = (float*)d_out;
    float* xout  = out;                    // B*N*20*3 = 61440
    float* cat   = out + 61440;            // B*N*N*4

    k_zero<<<256, 256, 0, stream>>>(WS);
    k_center<<<4, 256, 0, stream>>>(x, WS);
    k_dct_emb<<<4, 256, 0, stream>>>(x, vel, h, W_emb, b_emb, W_emb2, b_emb2, WS, HS + H_HH);
    k_coord<<<256, 256, 0, stream>>>(W_coord, W_vel, WS);
    k_xout<<<4, 256, 0, stream>>>(W_pred, WS, xout);

    k_swz<<<2, 256, 0, stream>>>(Wc1, HS + H_WC1, 4, 4, 64);
    k_swz<<<2, 256, 0, stream>>>(Wc2, HS + H_WC2, 2, 8, 128);
    k_swz<<<4, 256, 0, stream>>>(We1, HS + H_WE1, 8, 4, 64);
    k_swz<<<1, 256, 0, stream>>>(We2, HS + H_WE2, 2, 4, 64);
    k_swz<<<4, 256, 0, stream>>>(Wk1, HS + H_WK1, 8, 4, 64);
    k_swz<<<1, 64, 0, stream>>>(Wk2, HS + H_WK2, 2, 1, 4);

    k_edge1<<<BB * NN, 128, 0, stream>>>(WS + F_COORD, HS + H_HH,
        HS + H_WC1, HS + H_WC2, HS + H_WE1, HS + H_WE2,
        bc1, bc2, be1, be2, WS + F_AGG);

    k_node<<<4, 256, 0, stream>>>(Wn1, bn1, Wn2, bn2, WS, HS + H_NODE);

    k_edge2<<<BB * NN, 128, 0, stream>>>(WS + F_COORD, HS + H_NODE,
        HS + H_WC1, HS + H_WC2, HS + H_WK1, HS + H_WK2,
        bc1, bc2, bk1, bk2, cat);
}
